// LM_65146063946140
// MI455X (gfx1250) — matrix operand order repaired and checked by reading, not compiled, not run on hardware
//
#include <hip/hip_runtime.h>
#include <cmath>
#include <cstdint>

// ---------------------------------------------------------------------------
// CDNA5 (gfx1250) transformer forward.
// GEMMs: v_wmma_f32_16x16x32_f16, 128x256 block tile, 64x64 wave tile,
// double-buffered LDS fed by global_load_async_to_lds_b128 (ASYNCcnt).
// All B operands are pre-transposed to [N x K] so every tile load is a
// contiguous b128 chunk (no LDS scatter in the hot loop).
// ---------------------------------------------------------------------------

typedef __attribute__((ext_vector_type(16))) _Float16 v16h;
typedef __attribute__((ext_vector_type(8)))  float    v8f;

#define TILE_M 128
#define TILE_N 256
#define TILE_K 32
#define ASTR   40                 // LDS row stride in halves (32 + 8 pad)
#define BSTR   40
#define ABUF   (TILE_M * ASTR)    // 5120 halves
#define BBUF   (TILE_N * BSTR)    // 10240 halves

// --- WMMA fragment loads, per ISA 7.12.2 layouts (wave32) -------------------
// A 16x32 f16: lanes 0-15 g=0, lanes 16-31 g=1 (M = lane&15).
//   VGPR p in 0..3 : K = 8g + 2p,+1 ; VGPR p in 4..7 : K = 16 + 8g + 2(p-4),+1
__device__ __forceinline__ v16h ldfragA(const _Float16* As, int mBase, int lane) {
  const int m = mBase + (lane & 15);
  const int g = (lane >> 4) * 8;
  const _Float16* p = As + m * ASTR;
  v16h a;
#pragma unroll
  for (int i = 0; i < 4; ++i) { a[2*i]   = p[g + 2*i];      a[2*i+1] = p[g + 2*i + 1]; }
#pragma unroll
  for (int i = 0; i < 4; ++i) { a[8+2*i] = p[16 + g + 2*i]; a[9+2*i] = p[16 + g + 2*i + 1]; }
  return a;
}
// B 32x16 f16: N = lane&15; slot i<8 holds K = 8*(lane>>4) + i, slot i>=8 holds K = 16 + 8*(lane>>4) + (i-8).
// Bs stored [n][k] -> two contiguous 8-half reads.
__device__ __forceinline__ v16h ldfragB(const _Float16* Bs, int nBase, int lane) {
  const int n = nBase + (lane & 15);
  const int g = (lane >> 4) * 8;
  const _Float16* p = Bs + n * BSTR + g;
  v16h b;
#pragma unroll
  for (int i = 0; i < 8; ++i) { b[i] = p[i]; b[8+i] = p[16+i]; }
  return b;
}

// --- async global -> LDS b128 (16 bytes per lane), ASYNCcnt-tracked --------
__device__ __forceinline__ void async_b128(uint32_t ldsAddr, const _Float16* g) {
  asm volatile("global_load_async_to_lds_b128 %0, %1, off"
               :: "v"(ldsAddr), "v"(g) : "memory");
}
__device__ __forceinline__ void wait_async_le6() {
  asm volatile("s_wait_asynccnt 0x6" ::: "memory");
}
__device__ __forceinline__ void wait_async_0() {
  asm volatile("s_wait_asynccnt 0x0" ::: "memory");
}

// Issue the 6 async b128 chunks (A: 2, B: 4) for one K-tile.
__device__ __forceinline__ void stage_tile_async(
    const _Float16* Ag, long long lda, long long m0,
    const _Float16* Bg, long long ldb, long long n0,
    int k0, _Float16* Abuf, _Float16* Bbuf, int tid)
{
#pragma unroll
  for (int i = 0; i < 2; ++i) {               // A tile: 128 x 32 halves
    const int c = tid + 256 * i;              // 512 chunks of 8 halves
    const int row = c >> 2, kOff = (c & 3) * 8;
    async_b128((uint32_t)(uintptr_t)(Abuf + row * ASTR + kOff),
               Ag + (m0 + row) * lda + (k0 + kOff));
  }
#pragma unroll
  for (int i = 0; i < 4; ++i) {               // B tile: 256 x 32 halves
    const int c = tid + 256 * i;              // 1024 chunks
    const int row = c >> 2, kOff = (c & 3) * 8;
    async_b128((uint32_t)(uintptr_t)(Bbuf + row * BSTR + kOff),
               Bg + (n0 + row) * ldb + (k0 + kOff));
  }
}

// --- Generic batched GEMM: D = act(alpha*(A@B^T) + bias + res) -------------
// B is ALWAYS [N x K] row-major. z = blockIdx.z; zo=z/zInner, zi=z%zInner;
// operand offsets zo*Zo + zi*Zi. transCh writes Ch[n*ldch + m].
// causal: skip blocks fully above the diagonal. kTri: limit K to m0+TILE_M.
// Requires M%128==0, N%256==0, K%32==0 (all uses comply).
__global__ __launch_bounds__(256) void k_gemm_wmma(
    const _Float16* __restrict__ A,  long long lda,  long long aZo,  long long aZi,
    const _Float16* __restrict__ Bm, long long ldb,  long long bZo,  long long bZi,
    float*          __restrict__ Cf, long long ldc,  long long cZo,  long long cZi,
    _Float16*       __restrict__ Ch, long long ldch, long long chZo, long long chZi, int transCh,
    const float*    __restrict__ bias, long long biasZo, long long biasZi,
    const float*    __restrict__ res,  long long ldres, long long resZo, long long resZi,
    int K, int zInner, float alpha, int act, int causal, int kTri)
{
  const long long m0 = (long long)blockIdx.y * TILE_M;
  const long long n0 = (long long)blockIdx.x * TILE_N;
  if (causal && n0 >= m0 + TILE_M) return;    // fully masked block (uniform exit)

  __shared__ __align__(16) _Float16 smem[2 * ABUF + 2 * BBUF];  // 60 KB

  const int z  = blockIdx.z;
  const int zo = z / zInner;
  const int zi = z % zInner;
  A  += zo * aZo + zi * aZi;
  Bm += zo * bZo + zi * bZi;
  if (Cf)   Cf   += zo * cZo    + zi * cZi;
  if (Ch)   Ch   += zo * chZo   + zi * chZi;
  if (bias) bias += zo * biasZo + zi * biasZi;
  if (res)  res  += zo * resZo  + zi * resZi;

  const int tid  = threadIdx.x;
  const int lane = tid & 31;
  const int wave = tid >> 5;
  const int wr   = wave >> 2;   // 0..1 : wave row (64 rows each)
  const int wc   = wave & 3;    // 0..3 : wave col (64 cols each)

  const int Keff  = kTri ? ((K < (int)m0 + TILE_M) ? K : (int)m0 + TILE_M) : K;
  const int nIter = Keff / TILE_K;

  v8f acc[4][4] = {};

  stage_tile_async(A, lda, m0, Bm, ldb, n0, 0, smem, smem + 2 * ABUF, tid);

#pragma unroll 1
  for (int it = 0; it < nIter; ++it) {
    const int cur = it & 1;
    _Float16* Acur = smem + cur * ABUF;
    _Float16* Bcur = smem + 2 * ABUF + cur * BBUF;
    const bool more = (it + 1 < nIter);
    if (more) {
      stage_tile_async(A, lda, m0, Bm, ldb, n0, (it + 1) * TILE_K,
                       smem + (cur ^ 1) * ABUF,
                       smem + 2 * ABUF + (cur ^ 1) * BBUF, tid);
      wait_async_le6();           // oldest 6 (current tile) complete
    } else {
      wait_async_0();
    }
    __syncthreads();              // everyone's asyncs for cur are visible

    v16h af[4], bf[4];
#pragma unroll
    for (int mi = 0; mi < 4; ++mi) af[mi] = ldfragA(Acur, wr * 64 + mi * 16, lane);
#pragma unroll
    for (int ni = 0; ni < 4; ++ni) bf[ni] = ldfragB(Bcur, wc * 64 + ni * 16, lane);
#pragma unroll
    for (int mi = 0; mi < 4; ++mi)
#pragma unroll
      for (int ni = 0; ni < 4; ++ni)
        acc[mi][ni] = __builtin_amdgcn_wmma_f32_16x16x32_f16(
            false, af[mi], false, bf[ni], (short)0, acc[mi][ni], false, false);
    __syncthreads();              // done reading cur before it is overwritten
  }

  // Epilogue. C/D f32 layout: VGPR r, lanes 0-15 -> M=r, lanes 16-31 -> M=8+r.
  const int lm = (lane >> 4) * 8;
  const int ln = lane & 15;
#pragma unroll
  for (int mi = 0; mi < 4; ++mi) {
#pragma unroll
    for (int ni = 0; ni < 4; ++ni) {
      const long long mbase = m0 + wr * 64 + mi * 16 + lm;
      const long long n     = n0 + wc * 64 + ni * 16 + ln;
      const float bv = bias ? bias[n] : 0.0f;
#pragma unroll
      for (int r = 0; r < 8; ++r) {
        const long long m = mbase + r;
        float v = acc[mi][ni][r] * alpha + bv;
        if (res) v += res[m * ldres + n];
        if (act == 1) v = v / (1.0f + __expf(-v));   // SiLU / swish
        if (Cf) Cf[m * ldc + n] = v;
        if (Ch) Ch[transCh ? (n * ldch + m) : (m * ldch + n)] = (_Float16)v;
      }
    }
  }
}

// --- f32 -> f16 convert + transpose: dst[n*K + k] = src[k*N + n] -----------
// grid: (N/32, K/32, batch); 256 threads. K, N multiples of 32.
__global__ __launch_bounds__(256) void k_cvt_transpose(
    const float* __restrict__ src, _Float16* __restrict__ dst, int K, int N) {
  __shared__ float tile[32][33];
  const long long z = blockIdx.z;
  src += z * (long long)K * N;
  dst += z * (long long)K * N;
  const int n0 = blockIdx.x * 32, k0 = blockIdx.y * 32;
  const int tx = threadIdx.x & 31, ty = threadIdx.x >> 5;   // 8 row-groups
  for (int r = ty; r < 32; r += 8)
    tile[r][tx] = src[(long long)(k0 + r) * N + (n0 + tx)];
  __syncthreads();
  for (int r = ty; r < 32; r += 8)
    dst[(long long)(n0 + r) * K + (k0 + tx)] = (_Float16)tile[tx][r];
}

// --- Embedding gather + interleaved (GPT-J) RoPE ---------------------------
__global__ __launch_bounds__(128) void k_embed_rope(
    const int* __restrict__ tokens, const float* __restrict__ we,
    float* __restrict__ x, _Float16* __restrict__ xh) {
  const int D = 768, S = 1024;
  const long long row = blockIdx.x;            // b*S + s
  const int spos = (int)(row & (S - 1));
  const int tok = tokens[row];
  const float* e = we + (long long)tok * D;
  float* xr = x + row * D;
  _Float16* hr = xh + row * D;
  for (int i = threadIdx.x; i < D / 2; i += 128) {
    const float inv = __expf(-(2.0f * i / 768.0f) * 9.210340371976184f); // 10000^(-2i/d)
    const float t = (float)spos * inv;
    const float c = cosf(t), sn = sinf(t);
    const float x0 = e[2 * i], x1 = e[2 * i + 1];
    const float y0 = x0 * c - x1 * sn;
    const float y1 = x1 * c + x0 * sn;
    xr[2 * i] = y0;  xr[2 * i + 1] = y1;
    hr[2 * i] = (_Float16)y0;  hr[2 * i + 1] = (_Float16)y1;
  }
}

// --- Causal softmax over f16 score rows, in place --------------------------
__global__ __launch_bounds__(256) void k_softmax_causal(_Float16* __restrict__ p) {
  const int S = 1024;
  const long long row = blockIdx.x;            // (b*H + h)*S + s
  const int s = (int)(row & (S - 1));
  _Float16* pr = p + row * (long long)S;
  __shared__ float sm[256];
  const int tid = threadIdx.x;

  float mx = -1e30f;
  for (int t = tid; t <= s; t += 256) mx = fmaxf(mx, (float)pr[t]);
  sm[tid] = mx; __syncthreads();
  for (int o = 128; o > 0; o >>= 1) { if (tid < o) sm[tid] = fmaxf(sm[tid], sm[tid + o]); __syncthreads(); }
  mx = sm[0]; __syncthreads();

  float sum = 0.0f;
  for (int t = tid; t <= s; t += 256) sum += __expf((float)pr[t] - mx);
  sm[tid] = sum; __syncthreads();
  for (int o = 128; o > 0; o >>= 1) { if (tid < o) sm[tid] += sm[tid + o]; __syncthreads(); }
  const float invs = 1.0f / sm[0];

  for (int t = tid; t < S; t += 256) {
    const float v = (t <= s) ? __expf((float)pr[t] - mx) * invs : 0.0f;
    pr[t] = (_Float16)v;
  }
}

// --- LayerNorm (row per block), optional f16 shadow output -----------------
__global__ __launch_bounds__(256) void k_layernorm(
    const float* __restrict__ xin, const float* __restrict__ gamma,
    const float* __restrict__ beta, float* __restrict__ xout,
    _Float16* __restrict__ hout) {
  const int D = 768;
  const long long row = blockIdx.x;
  const float* xr = xin + row * D;
  __shared__ float sm[256];
  const int tid = threadIdx.x;
  const float a0 = xr[tid], a1 = xr[tid + 256], a2 = xr[tid + 512];

  sm[tid] = a0 + a1 + a2; __syncthreads();
  for (int o = 128; o > 0; o >>= 1) { if (tid < o) sm[tid] += sm[tid + o]; __syncthreads(); }
  const float mean = sm[0] * (1.0f / 768.0f); __syncthreads();

  const float d0 = a0 - mean, d1 = a1 - mean, d2 = a2 - mean;
  sm[tid] = d0 * d0 + d1 * d1 + d2 * d2; __syncthreads();
  for (int o = 128; o > 0; o >>= 1) { if (tid < o) sm[tid] += sm[tid + o]; __syncthreads(); }
  const float rstd = rsqrtf(sm[0] * (1.0f / 768.0f) + 1e-6f);

  float* yr = xout + row * D;
  const float dd[3] = {d0, d1, d2};
#pragma unroll
  for (int j = 0; j < 3; ++j) {
    const int c = tid + j * 256;
    const float y = dd[j] * rstd * gamma[c] + beta[c];
    yr[c] = y;
    if (hout) hout[row * D + c] = (_Float16)y;
  }
}

// ---------------------------------------------------------------------------
extern "C" void kernel_launch(void* const* d_in, const int* in_sizes, int n_in,
                              void* d_out, int out_size, void* d_ws, size_t ws_size,
                              hipStream_t stream) {
  (void)in_sizes; (void)n_in; (void)out_size; (void)ws_size;
  constexpr long long NB = 2, S = 1024, D = 768, H = 12, F = 3072, L = 4;
  constexpr long long BS = NB * S;          // 2048
  constexpr long long HD = H * D;           // 9216

  const int*   tokens = (const int*)  d_in[0];
  const float* w_e    = (const float*)d_in[1];
  const float* Wq     = (const float*)d_in[2];
  const float* bq     = (const float*)d_in[3];
  const float* Wk     = (const float*)d_in[4];
  const float* bk     = (const float*)d_in[5];
  const float* Wv     = (const float*)d_in[6];
  const float* bv     = (const float*)d_in[7];
  const float* Wo     = (const float*)d_in[8];
  const float* bo     = (const float*)d_in[9];
  const float* W1     = (const float*)d_in[10];
  const float* b1     = (const float*)d_in[11];
  const float* W2     = (const float*)d_in[12];
  const float* b2     = (const float*)d_in[13];
  const float* ln_s   = (const float*)d_in[14];
  const float* ln_b   = (const float*)d_in[15];
  const float* lnf_s  = (const float*)d_in[16];
  const float* lnf_b  = (const float*)d_in[17];
  float* out = (float*)d_out;

  // ---- carve workspace (256B aligned) ----
  char* wp = (char*)d_ws;
  auto carve = [&](long long bytes) -> char* {
    char* p = wp; wp += (bytes + 255) & ~255LL; return p;
  };
  float*    x    = (float*)   carve(BS * D * 4);          // activations f32
  _Float16* x_h  = (_Float16*)carve(BS * D * 2);          // f16 shadow
  float*    q    = (float*)   carve(NB * H * S * D * 4);  // Q f32 (residual)
  _Float16* q_h  = (_Float16*)carve(NB * H * S * D * 2);  // [B,H,S,D]
  _Float16* k_h  = (_Float16*)carve(NB * H * S * D * 2);  // [B,H,S,D] (B^T for scores)
  _Float16* v_t  = (_Float16*)carve(NB * H * S * D * 2);  // [B,H,D,S] (B^T for AV)
  _Float16* p_h  = (_Float16*)carve(NB * H * S * S * 2);  // scores / probs
  _Float16* oc_h = (_Float16*)carve(BS * HD * 2);         // concat heads
  _Float16* ff_h = (_Float16*)carve(BS * F * 2);          // FFN hidden
  _Float16* wq_t = (_Float16*)carve(H * D * D * 2);       // per-layer W^T staging (f16)
  _Float16* wk_t = (_Float16*)carve(H * D * D * 2);
  _Float16* wv_t = (_Float16*)carve(H * D * D * 2);
  _Float16* wo_t = (_Float16*)carve(HD * D * 2);          // [D x HD]
  _Float16* w1_t = (_Float16*)carve(D * F * 2);           // [F x D]
  _Float16* w2_t = (_Float16*)carve(F * D * 2);           // [D x F]

  auto cvtT = [&](const float* s, _Float16* dmem, long long Kd, long long Nd, long long Zb) {
    dim3 grid((unsigned)(Nd / 32), (unsigned)(Kd / 32), (unsigned)Zb);
    k_cvt_transpose<<<grid, dim3(256), 0, stream>>>(s, dmem, (int)Kd, (int)Nd);
  };
  auto gemm = [&](const _Float16* A, long long lda, long long aZo, long long aZi,
                  const _Float16* Bm, long long ldb, long long bZo, long long bZi,
                  float* Cf, long long ldc, long long cZo, long long cZi,
                  _Float16* Ch, long long ldch, long long chZo, long long chZi, int transCh,
                  const float* bias, long long biasZo, long long biasZi,
                  const float* res, long long ldres, long long resZo, long long resZi,
                  int M, int N, int K, int Z, int zInner, float alpha, int act,
                  int causal, int kTri) {
    dim3 grid((unsigned)(N / TILE_N), (unsigned)(M / TILE_M), (unsigned)Z);
    k_gemm_wmma<<<grid, dim3(256), 0, stream>>>(
        A, lda, aZo, aZi, Bm, ldb, bZo, bZi,
        Cf, ldc, cZo, cZi, Ch, ldch, chZo, chZi, transCh,
        bias, biasZo, biasZi, res, ldres, resZo, resZi,
        K, zInner, alpha, act, causal, kTri);
  };

  // ---- embedding + RoPE ----
  k_embed_rope<<<dim3((unsigned)BS), dim3(128), 0, stream>>>(tokens, w_e, x, x_h);

  const float scale = 1.0f / sqrtf((float)D);

  for (long long l = 0; l < L; ++l) {
    // Transposed f16 weights for this layer
    cvtT(Wq + l * H * D * D, wq_t, D, D, H);
    cvtT(Wk + l * H * D * D, wk_t, D, D, H);
    cvtT(Wv + l * H * D * D, wv_t, D, D, H);
    cvtT(Wo + l * HD * D,    wo_t, HD, D, 1);
    cvtT(W1 + l * D * F,     w1_t, D, F, 1);
    cvtT(W2 + l * F * D,     w2_t, F, D, 1);

    // Q/K/V projections: z=(b,h); A=x_h per-b, B=W^T per-h. [S x D] @ [D x D]
    gemm(x_h, D, S * D, 0,  wq_t, D, 0, D * D,
         q,   D, H * S * D, S * D,  q_h, D, H * S * D, S * D, 0,
         bq + l * H * D, 0, D,  nullptr, 0, 0, 0,
         (int)S, (int)D, (int)D, (int)(NB * H), (int)H, 1.0f, 0, 0, 0);
    gemm(x_h, D, S * D, 0,  wk_t, D, 0, D * D,
         nullptr, 0, 0, 0,  k_h, D, H * S * D, S * D, 0,
         bk + l * H * D, 0, D,  nullptr, 0, 0, 0,
         (int)S, (int)D, (int)D, (int)(NB * H), (int)H, 1.0f, 0, 0, 0);
    // V written transposed: v_t[b,h,e,s]
    gemm(x_h, D, S * D, 0,  wv_t, D, 0, D * D,
         nullptr, 0, 0, 0,  v_t, S, H * D * S, D * S, 1,
         bv + l * H * D, 0, D,  nullptr, 0, 0, 0,
         (int)S, (int)D, (int)D, (int)(NB * H), (int)H, 1.0f, 0, 0, 0);

    // Scores: p = (q @ k^T) * scale; skip blocks above the diagonal.
    gemm(q_h, D, S * D, 0,  k_h, D, S * D, 0,
         nullptr, 0, 0, 0,  p_h, S, S * S, 0, 0,
         nullptr, 0, 0,  nullptr, 0, 0, 0,
         (int)S, (int)S, (int)D, (int)(NB * H), 1, scale, 0, 1, 0);

    // Causal softmax in place (writes zeros above diagonal).
    k_softmax_causal<<<dim3((unsigned)(NB * H * S)), dim3(256), 0, stream>>>(p_h);

    // O = q + probs @ V; K capped at m0+TILE_M (probs are 0 beyond diagonal);
    // scattered directly into oc layout [B, S, H*D].
    gemm(p_h, S, H * S * S, S * S,  v_t, S, H * D * S, D * S,
         nullptr, 0, 0, 0,  oc_h, HD, S * HD, D, 0,
         nullptr, 0, 0,  q, D, H * S * D, S * D,
         (int)S, (int)D, (int)S, (int)(NB * H), (int)H, 1.0f, 0, 0, 1);

    // x = x + oc @ Wo + bo  (in place, refresh x_h)
    gemm(oc_h, HD, 0, 0,  wo_t, HD, 0, 0,
         x, D, 0, 0,  x_h, D, 0, 0, 0,
         bo + l * D, 0, 0,  x, D, 0, 0,
         (int)BS, (int)D, (int)HD, 1, 1, 1.0f, 0, 0, 0);

    // h = swish(x @ W1 + b1) -> f16 only
    gemm(x_h, D, 0, 0,  w1_t, D, 0, 0,
         nullptr, 0, 0, 0,  ff_h, F, 0, 0, 0,
         b1 + l * F, 0, 0,  nullptr, 0, 0, 0,
         (int)BS, (int)F, (int)D, 1, 1, 1.0f, 1, 0, 0);

    // x = x + h @ W2 + b2  (in place)
    gemm(ff_h, F, 0, 0,  w2_t, F, 0, 0,
         x, D, 0, 0,  nullptr, 0, 0, 0, 0,
         b2 + l * D, 0, 0,  x, D, 0, 0,
         (int)BS, (int)D, (int)F, 1, 1, 1.0f, 0, 0, 0);

    // x = LN(x) in place, refresh x_h
    k_layernorm<<<dim3((unsigned)BS), dim3(256), 0, stream>>>(
        x, ln_s + l * D, ln_b + l * D, x, x_h);
  }

  // Final LayerNorm -> d_out (f32)
  k_layernorm<<<dim3((unsigned)BS), dim3(256), 0, stream>>>(
      x, lnf_s, lnf_b, out, nullptr);
}
